// SparseTernaryViTFeedForward_41841571398455
// MI455X (gfx1250) — matrix operand order repaired and checked by reading, not compiled, not run on hardware
//
#include <hip/hip_runtime.h>
#include <hip/hip_bf16.h>

typedef __attribute__((ext_vector_type(16))) _Float16 v16h;
typedef __attribute__((ext_vector_type(8)))  _Float16 v8h;
typedef __attribute__((ext_vector_type(8)))  float    v8f;

#define NB 2048
#define HRANGE 0.25f
#define FIXSCALE 16777216.0f   // 2^24 fixed-point for deterministic |W| sum

#define USE_ASYNC_LDS 1

// ---------------------------------------------------------------------------
// Block reduction helper (256 threads)
// ---------------------------------------------------------------------------
__device__ inline float block_reduce_sum(float v, float* red) {
    const int t = threadIdx.x;
    red[t] = v;
    __syncthreads();
    for (int off = 128; off > 0; off >>= 1) {
        if (t < off) red[t] += red[t + off];
        __syncthreads();
    }
    float r = red[0];
    __syncthreads();
    return r;
}

__device__ inline float gelu_tanh(float x) {
    const float c = 0.7978845608028654f;  // sqrt(2/pi)
    float x3 = x * x * x;
    return 0.5f * x * (1.0f + tanhf(c * (x + 0.044715f * x3)));
}

// ---------------------------------------------------------------------------
// 1) |W| histogram + fixed-point sum (deterministic)
// ---------------------------------------------------------------------------
__global__ __launch_bounds__(256) void absstats_kernel(
    const float* __restrict__ W, int n,
    unsigned int* __restrict__ hist, unsigned long long* __restrict__ fixsum) {
    __shared__ unsigned int hsh[NB];
    for (int i = threadIdx.x; i < NB; i += 256) hsh[i] = 0u;
    __syncthreads();

    unsigned long long local = 0ull;
    for (int i = blockIdx.x * 256 + threadIdx.x; i < n; i += gridDim.x * 256) {
        float a = fabsf(W[i]);
        local += (unsigned long long)(a * FIXSCALE + 0.5f);
        int bin = (int)(a * ((float)NB / HRANGE));
        if (bin > NB - 1) bin = NB - 1;
        atomicAdd(&hsh[bin], 1u);
    }
    __syncthreads();
    for (int i = threadIdx.x; i < NB; i += 256)
        if (hsh[i]) atomicAdd(&hist[i], hsh[i]);
    atomicAdd(fixsum, local);
}

// ---------------------------------------------------------------------------
// 2) threshold (approx quantile) + scale (mean |W|)
// ---------------------------------------------------------------------------
__global__ void findthresh_kernel(
    const unsigned int* __restrict__ hist1, const unsigned long long* __restrict__ sum1,
    const unsigned int* __restrict__ hist2, const unsigned long long* __restrict__ sum2,
    const float* __restrict__ sparsity, float* __restrict__ params, int n1, int n2) {
    if (threadIdx.x == 0 && blockIdx.x == 0) {
        float sp = *sparsity;
        {
            float target = sp * (float)n1;
            float cum = 0.0f;
            int bin = 0;
            for (bin = 0; bin < NB; ++bin) { cum += (float)hist1[bin]; if (cum >= target) break; }
            params[0] = (float)((double)*sum1 / (double)FIXSCALE / (double)n1);
            params[1] = ((float)bin + 0.5f) * (HRANGE / (float)NB);
        }
        {
            float target = sp * (float)n2;
            float cum = 0.0f;
            int bin = 0;
            for (bin = 0; bin < NB; ++bin) { cum += (float)hist2[bin]; if (cum >= target) break; }
            params[2] = (float)((double)*sum2 / (double)FIXSCALE / (double)n2);
            params[3] = ((float)bin + 0.5f) * (HRANGE / (float)NB);
        }
    }
}

// ---------------------------------------------------------------------------
// 3) ternary quantize W -> f16 {-scale, 0, +scale}; optional per-col gamma fold
// ---------------------------------------------------------------------------
__global__ __launch_bounds__(256) void quantize_kernel(
    const float* __restrict__ W, _Float16* __restrict__ q,
    const float* __restrict__ params, int which, int n) {
    float scale = params[which * 2 + 0];
    float thresh = params[which * 2 + 1];
    for (int i = blockIdx.x * 256 + threadIdx.x; i < n; i += gridDim.x * 256) {
        float w = W[i];
        float a = fabsf(w);
        float v = (a > thresh) ? (w > 0.0f ? scale : -scale) : 0.0f;
        q[i] = (_Float16)v;
    }
}

__global__ __launch_bounds__(256) void quantize_fold_kernel(
    const float* __restrict__ W, _Float16* __restrict__ q,
    const float* __restrict__ params, int which,
    const float* __restrict__ gamma, int ncols, int n) {
    float scale = params[which * 2 + 0];
    float thresh = params[which * 2 + 1];
    for (int i = blockIdx.x * 256 + threadIdx.x; i < n; i += gridDim.x * 256) {
        float w = W[i];
        float a = fabsf(w);
        float v = (a > thresh) ? (w > 0.0f ? scale : -scale) : 0.0f;
        q[i] = (_Float16)(v * gamma[i % ncols]);
    }
}

// ---------------------------------------------------------------------------
// 3b) per-output-column constants for the LN fold:
//     Sg[n] = sum_k gamma[k]*tern(W2[n,k]);  Sb[n] = sum_k beta[k]*tern(W2[n,k])
// ---------------------------------------------------------------------------
__global__ __launch_bounds__(256) void wsum_kernel(
    const float* __restrict__ W, const float* __restrict__ g, const float* __restrict__ b,
    const float* __restrict__ params, int which,
    float* __restrict__ Sg, float* __restrict__ Sb, int Kc) {
    __shared__ float red[256];
    const int n = blockIdx.x;
    float scale = params[which * 2 + 0];
    float thresh = params[which * 2 + 1];
    float sg = 0.0f, sb = 0.0f;
    for (int k = threadIdx.x; k < Kc; k += 256) {
        float w = W[(size_t)n * Kc + k];
        float t = (fabsf(w) > thresh) ? (w > 0.0f ? scale : -scale) : 0.0f;
        sg += g[k] * t;
        sb += b[k] * t;
    }
    sg = block_reduce_sum(sg, red);
    sb = block_reduce_sum(sb, red);
    if (threadIdx.x == 0) { Sg[n] = sg; Sb[n] = sb; }
}

// ---------------------------------------------------------------------------
// 4) LayerNorm (one block per row) -> f16
// ---------------------------------------------------------------------------
template <int PER, typename InT>
__global__ __launch_bounds__(256) void ln_kernel(
    const InT* __restrict__ x, const float* __restrict__ g, const float* __restrict__ b,
    _Float16* __restrict__ out, int D) {
    __shared__ float red[256];
    const int r = blockIdx.x;
    const InT* xr = x + (size_t)r * D;
    _Float16* orow = out + (size_t)r * D;

    float vals[PER];
    float s = 0.0f;
#pragma unroll
    for (int i = 0; i < PER; ++i) {
        float v = (float)xr[threadIdx.x + i * 256];
        vals[i] = v;
        s += v;
    }
    s = block_reduce_sum(s, red);
    float mu = s / (float)D;

    float s2 = 0.0f;
#pragma unroll
    for (int i = 0; i < PER; ++i) {
        float d = vals[i] - mu;
        s2 += d * d;
    }
    s2 = block_reduce_sum(s2, red);
    float rs = rsqrtf(s2 / (float)D + 1e-5f);

#pragma unroll
    for (int i = 0; i < PER; ++i) {
        int e = threadIdx.x + i * 256;
        orow[e] = (_Float16)((vals[i] - mu) * rs * g[e] + b[e]);
    }
}

// ---------------------------------------------------------------------------
// 4b) row stats only (mu, rsqrt(var+eps)) -- no write-back pass
// ---------------------------------------------------------------------------
template <int PER>
__global__ __launch_bounds__(256) void rowstats_kernel(
    const _Float16* __restrict__ x, float* __restrict__ rowMu, float* __restrict__ rowRs, int D) {
    __shared__ float red[256];
    const int r = blockIdx.x;
    const _Float16* xr = x + (size_t)r * D;

    float vals[PER];
    float s = 0.0f;
#pragma unroll
    for (int i = 0; i < PER; ++i) {
        float v = (float)xr[threadIdx.x + i * 256];
        vals[i] = v;
        s += v;
    }
    s = block_reduce_sum(s, red);
    float mu = s / (float)D;

    float s2 = 0.0f;
#pragma unroll
    for (int i = 0; i < PER; ++i) {
        float d = vals[i] - mu;
        s2 += d * d;
    }
    s2 = block_reduce_sum(s2, red);
    if (threadIdx.x == 0) {
        rowMu[r] = mu;
        rowRs[r] = rsqrtf(s2 / (float)D + 1e-5f);
    }
}

// ---------------------------------------------------------------------------
// 5) WMMA GEMM:  out[M,N] = A[M,K] (f16, row-major) * W[N,K]^T (f16) + epilogue
//    128x128 block tile, K-step 64, 8 waves (4x2), each wave 32x64.
//    Double-buffered LDS fed by GLOBAL_LOAD_ASYNC_TO_LDS_B128 (ASYNCcnt).
//    Epilogue: GELU (GEMM1) or folded LayerNorm (GEMM2).
// ---------------------------------------------------------------------------
union V16U { v16h v; v8h h[2]; };

#define LDS_STRIDE 72   // 64 halves + 8 pad (144 B = 36 banks: conflict-free)

__device__ __forceinline__ unsigned lds_off32(const void* p) {
    return (unsigned)(unsigned long long)p;  // low 32 bits = LDS byte offset
}

#if USE_ASYNC_LDS
__device__ __forceinline__ void async_copy_b128(unsigned lds, const _Float16* g) {
    asm volatile("global_load_async_to_lds_b128 %0, %1, off"
                 :: "v"(lds), "v"(g) : "memory");
}
__device__ __forceinline__ void wait_asynccnt0() {
    asm volatile("s_wait_asynccnt 0x0" ::: "memory");
}
#endif

template <bool GELU, bool OUT_F16, bool LNFOLD>
__global__ __launch_bounds__(256) void gemm_wmma_kernel(
    const _Float16* __restrict__ A, const _Float16* __restrict__ W,
    const float* __restrict__ bias, void* __restrict__ outp,
    const float* __restrict__ rowMu, const float* __restrict__ rowRs,
    const float* __restrict__ Sg, const float* __restrict__ Sb,
    int M, int N, int K) {
#if USE_ASYNC_LDS
    __shared__ __align__(16) _Float16 As[2][128 * LDS_STRIDE];
    __shared__ __align__(16) _Float16 Bs[2][128 * LDS_STRIDE];
#else
    __shared__ __align__(16) _Float16 As[1][128 * LDS_STRIDE];
    __shared__ __align__(16) _Float16 Bs[1][128 * LDS_STRIDE];
#endif

    const int tid   = threadIdx.x;
    const int lane  = tid & 31;
    const int wave  = tid >> 5;
    const int waveM = wave & 3;
    const int waveN = wave >> 2;
    const int mBase = blockIdx.y * 128;
    const int nBase = blockIdx.x * 128;
    const int row16 = lane & 15;
    const int sel   = lane >> 4;

    v8f acc[2][4];
#pragma unroll
    for (int mi = 0; mi < 2; ++mi)
#pragma unroll
        for (int ni = 0; ni < 4; ++ni) acc[mi][ni] = (v8f)(0.0f);

    const int T = K >> 6;

#if USE_ASYNC_LDS
#pragma unroll
    for (int i = 0; i < 4; ++i) {
        int chunk = tid + i * 256;
        int r  = chunk >> 3;
        int cc = (chunk & 7) * 8;
        async_copy_b128(lds_off32(&As[0][r * LDS_STRIDE + cc]),
                        A + (size_t)(mBase + r) * K + cc);
        async_copy_b128(lds_off32(&Bs[0][r * LDS_STRIDE + cc]),
                        W + (size_t)(nBase + r) * K + cc);
    }
    for (int t = 0; t < T; ++t) {
        const int cur = t & 1;
        wait_asynccnt0();
        __syncthreads();
        if (t + 1 < T) {
            const int k0 = (t + 1) << 6;
#pragma unroll
            for (int i = 0; i < 4; ++i) {
                int chunk = tid + i * 256;
                int r  = chunk >> 3;
                int cc = (chunk & 7) * 8;
                async_copy_b128(lds_off32(&As[cur ^ 1][r * LDS_STRIDE + cc]),
                                A + (size_t)(mBase + r) * K + k0 + cc);
                async_copy_b128(lds_off32(&Bs[cur ^ 1][r * LDS_STRIDE + cc]),
                                W + (size_t)(nBase + r) * K + k0 + cc);
            }
        }
#else
    uint4 ra[4], rb[4];
#pragma unroll
    for (int i = 0; i < 4; ++i) {
        int chunk = tid + i * 256;
        int r  = chunk >> 3;
        int cc = (chunk & 7) * 8;
        ra[i] = *(const uint4*)(A + (size_t)(mBase + r) * K + cc);
        rb[i] = *(const uint4*)(W + (size_t)(nBase + r) * K + cc);
    }
    for (int t = 0; t < T; ++t) {
        const int cur = 0;
        __syncthreads();
#pragma unroll
        for (int i = 0; i < 4; ++i) {
            int chunk = tid + i * 256;
            int r  = chunk >> 3;
            int cc = (chunk & 7) * 8;
            *(uint4*)(&As[0][r * LDS_STRIDE + cc]) = ra[i];
            *(uint4*)(&Bs[0][r * LDS_STRIDE + cc]) = rb[i];
        }
        __syncthreads();
        if (t + 1 < T) {
            const int k0 = (t + 1) << 6;
#pragma unroll
            for (int i = 0; i < 4; ++i) {
                int chunk = tid + i * 256;
                int r  = chunk >> 3;
                int cc = (chunk & 7) * 8;
                ra[i] = *(const uint4*)(A + (size_t)(mBase + r) * K + k0 + cc);
                rb[i] = *(const uint4*)(W + (size_t)(nBase + r) * K + k0 + cc);
            }
        }
#endif
#pragma unroll
        for (int c = 0; c < 2; ++c) {
            const int cb = c * 32;
            v16h af[2];
#pragma unroll
            for (int mi = 0; mi < 2; ++mi) {
                const _Float16* p = &As[cur][(waveM * 32 + mi * 16 + row16) * LDS_STRIDE + cb];
                V16U u;
                u.h[0] = *(const v8h*)(p + sel * 8);
                u.h[1] = *(const v8h*)(p + 16 + sel * 8);
                af[mi] = u.v;
            }
            v16h bf[4];
#pragma unroll
            for (int ni = 0; ni < 4; ++ni) {
                const _Float16* p = &Bs[cur][(waveN * 64 + ni * 16 + row16) * LDS_STRIDE + cb];
                V16U u;
                u.h[0] = *(const v8h*)(p + sel * 8);
                u.h[1] = *(const v8h*)(p + 16 + sel * 8);
                bf[ni] = u.v;
            }
#pragma unroll
            for (int mi = 0; mi < 2; ++mi)
#pragma unroll
                for (int ni = 0; ni < 4; ++ni)
                    acc[mi][ni] = __builtin_amdgcn_wmma_f32_16x16x32_f16(
                        false, af[mi], false, bf[ni], (short)0, acc[mi][ni], false, false);
        }
    }

    // Epilogue. C/D layout: vgpr i -> row = i + sel*8; col = lane&15
    if (LNFOLD) {
#pragma unroll
        for (int mi = 0; mi < 2; ++mi) {
            int rbase = mBase + waveM * 32 + mi * 16 + sel * 8;
#pragma unroll
            for (int i = 0; i < 8; ++i) {
                int row = rbase + i;
                float mu = rowMu[row];
                float rs = rowRs[row];
#pragma unroll
                for (int ni = 0; ni < 4; ++ni) {
                    int gcol = nBase + waveN * 64 + ni * 16 + row16;
                    float v = rs * acc[mi][ni][i] - rs * mu * Sg[gcol] + Sb[gcol] + bias[gcol];
                    ((float*)outp)[(size_t)row * N + gcol] = v;
                }
            }
        }
    } else {
#pragma unroll
        for (int mi = 0; mi < 2; ++mi) {
#pragma unroll
            for (int ni = 0; ni < 4; ++ni) {
                int gcol = nBase + waveN * 64 + ni * 16 + row16;
                int rbase = mBase + waveM * 32 + mi * 16 + sel * 8;
                float bv = bias[gcol];
#pragma unroll
                for (int i = 0; i < 8; ++i) {
                    float v = acc[mi][ni][i] + bv;
                    if (GELU) v = gelu_tanh(v);
                    size_t idx = (size_t)(rbase + i) * N + gcol;
                    if (OUT_F16)
                        ((_Float16*)outp)[idx] = (_Float16)v;
                    else
                        ((float*)outp)[idx] = v;
                }
            }
        }
    }
}

// ---------------------------------------------------------------------------
// launch
// ---------------------------------------------------------------------------
extern "C" void kernel_launch(void* const* d_in, const int* in_sizes, int n_in,
                              void* d_out, int out_size, void* d_ws, size_t ws_size,
                              hipStream_t stream) {
    const float* x      = (const float*)d_in[0];
    const float* ln1_g  = (const float*)d_in[1];
    const float* ln1_b  = (const float*)d_in[2];
    const float* W1     = (const float*)d_in[3];
    const float* b1     = (const float*)d_in[4];
    const float* ln2_g  = (const float*)d_in[5];
    const float* ln2_b  = (const float*)d_in[6];
    const float* W2     = (const float*)d_in[7];
    const float* b2     = (const float*)d_in[8];
    const float* spars  = (const float*)d_in[9];

    const int D = in_sizes[1];                  // 768
    const int H = in_sizes[5];                  // 3072
    const long M = (long)in_sizes[0] / D;       // 32768 rows
    const int n1 = H * D;
    const int n2 = D * H;

    // workspace layout
    char* ws = (char*)d_ws;
    unsigned int*       hist1 = (unsigned int*)(ws + 0);
    unsigned int*       hist2 = (unsigned int*)(ws + 8192);
    unsigned long long* sum1  = (unsigned long long*)(ws + 16384);
    unsigned long long* sum2  = (unsigned long long*)(ws + 16392);
    float*              params = (float*)(ws + 16400);
    size_t off = 16640;
    _Float16* qW1  = (_Float16*)(ws + off); off += ((size_t)n1 * 2 + 255) & ~255ull;
    _Float16* qW2  = (_Float16*)(ws + off); off += ((size_t)n2 * 2 + 255) & ~255ull;
    _Float16* A1   = (_Float16*)(ws + off); off += ((size_t)M * D * 2 + 255) & ~255ull;
    _Float16* H1   = (_Float16*)(ws + off); off += ((size_t)M * H * 2 + 255) & ~255ull;
    float*    rowMu = (float*)(ws + off);   off += ((size_t)M * 4 + 255) & ~255ull;
    float*    rowRs = (float*)(ws + off);   off += ((size_t)M * 4 + 255) & ~255ull;
    float*    Sg    = (float*)(ws + off);   off += ((size_t)D * 4 + 255) & ~255ull;
    float*    Sb    = (float*)(ws + off);   off += ((size_t)D * 4 + 255) & ~255ull;
    (void)ws_size;

    hipMemsetAsync(d_ws, 0, 16640, stream);

    absstats_kernel<<<1024, 256, 0, stream>>>(W1, n1, hist1, sum1);
    absstats_kernel<<<1024, 256, 0, stream>>>(W2, n2, hist2, sum2);
    findthresh_kernel<<<1, 32, 0, stream>>>(hist1, sum1, hist2, sum2, spars, params, n1, n2);

    // qW1 = ternary(W1); qW2 = ternary(W2) * gamma2[k]  (LN2 gamma folded in)
    quantize_kernel<<<2048, 256, 0, stream>>>(W1, qW1, params, 0, n1);
    quantize_fold_kernel<<<2048, 256, 0, stream>>>(W2, qW2, params, 1, ln2_g, H, n2);

    // per-column LN-fold constants for GEMM2
    wsum_kernel<<<D, 256, 0, stream>>>(W2, ln2_g, ln2_b, params, 1, Sg, Sb, H);

    // LN1: f32 x -> f16 A1
    ln_kernel<3, float><<<(int)M, 256, 0, stream>>>(x, ln1_g, ln1_b, A1, D);

    // GEMM1 + bias + GELU -> f16 H1   (A1[M,D] * qW1[H,D]^T)
    gemm_wmma_kernel<true, true, false><<<dim3(H / 128, (int)(M / 128)), 256, 0, stream>>>(
        A1, qW1, b1, (void*)H1, nullptr, nullptr, nullptr, nullptr, (int)M, H, D);

    // LN2 stats only (no write-back of normalized activations)
    rowstats_kernel<12><<<(int)M, 256, 0, stream>>>(H1, rowMu, rowRs, H);

    // GEMM2 with folded LN2 + bias -> f32 out   (H1[M,H] * qW2[D,H]^T)
    gemm_wmma_kernel<false, false, true><<<dim3(D / 128, (int)(M / 128)), 256, 0, stream>>>(
        H1, qW2, b2, d_out, rowMu, rowRs, Sg, Sb, (int)M, D, H);
}
